// MultiHeadAttention_19765439496847
// MI455X (gfx1250) — matrix operand order repaired and checked by reading, not compiled, not run on hardware
//
#include <hip/hip_runtime.h>
#include <stdint.h>

// ---------------------------------------------------------------------------
// MI455X (gfx1250) MHA forward. fp32 -> bf16 WMMA (f32 accum) -> fp32.
// v2: LDS tiling fed by async load-to-LDS (ASYNCcnt), double buffered, so
// WMMA bursts overlap the async fills instead of stalling on loadcnt.
// ---------------------------------------------------------------------------

typedef __attribute__((ext_vector_type(16))) __bf16 v16bf;
typedef __attribute__((ext_vector_type(8)))  __bf16 v8bf;
typedef __attribute__((ext_vector_type(4)))  __bf16 v4bf;
typedef __attribute__((ext_vector_type(8)))  float  v8f;
typedef __attribute__((ext_vector_type(4)))  float  v4f;

#define DEVINL static __device__ __forceinline__

constexpr int N_ = 4, T_ = 2048, E_ = 768, H_ = 12, D_ = 64;
constexpr int NH = N_ * H_;   // 48
constexpr int NT = N_ * T_;   // 8192

// ---- CDNA5 async/wait primitives (inline asm: portable across toolchains) --
DEVINL void async_b128(const __bf16* g, const __bf16* l) {
  // VDST = per-lane LDS byte address (low 32 bits of the flat LDS address),
  // VADDR = 64-bit global address.  Tracked by ASYNCcnt.
  asm volatile("global_load_async_to_lds_b128 %0, %1, off"
               :: "v"((uint32_t)(uintptr_t)l), "v"((uint64_t)(uintptr_t)g)
               : "memory");
}
DEVINL void wait_async0() { asm volatile("s_wait_asynccnt 0" ::: "memory"); }
DEVINL void wait_ds0()    { asm volatile("s_wait_dscnt 0"    ::: "memory"); }

// ---- WMMA fragment helpers -------------------------------------------------
DEVINL v16bf combine16(v8bf lo, v8bf hi) {
  v16bf a;
#pragma unroll
  for (int i = 0; i < 8; ++i) { a[i] = lo[i]; a[8 + i] = hi[i]; }
  return a;
}
// A-fragment 16x32: lane {row=l&15, half=l>>4}; 16 K-contiguous elems,
// K=half*16+0..15.
DEVINL v16bf make_a_frag(const __bf16* row, int kbase, int half) {
  const __bf16* p = row + kbase + half * 16;
  return combine16(*(const v8bf*)(p), *(const v8bf*)(p + 8));
}
// B-fragment 32x16: lane {col=l&15, half=l>>4}; 16 K-contiguous elems.
// Loaded as 2x16B so 16B-aligned (padded LDS rows) addresses are legal.
DEVINL v16bf make_b_frag(const __bf16* p) {
  return combine16(*(const v8bf*)(p), *(const v8bf*)(p + 8));
}
DEVINL v8f wmma_bf16(v16bf a, v16bf b, v8f c) {
  return __builtin_amdgcn_wmma_f32_16x16x32_bf16(false, a, false, b,
                                                 (short)0, c, false, false);
}

// ---------------------------------------------------------------------------
// fp32 -> bf16 conversion (vec4)
// ---------------------------------------------------------------------------
__global__ __launch_bounds__(256) void cvt_f32_bf16(const float* __restrict__ in,
                                                    __bf16* __restrict__ out, int n4) {
  int i = blockIdx.x * 256 + threadIdx.x;
  if (i < n4) {
    v4f f = ((const v4f*)in)[i];
    v4bf o;
#pragma unroll
    for (int j = 0; j < 4; ++j) o[j] = (__bf16)f[j];
    ((v4bf*)out)[i] = o;
  }
}

// ---------------------------------------------------------------------------
// QKV projection: Y = x @ W^T, W in {Wq,Wk,Wv}.
// Block = 128x128 tile; 8 waves = 4(M) x 2(N); wave = 32x64 (8 accums).
// K chunks of 64 staged in LDS via async b128 loads, double buffered.
// Q,K -> [N,H,T,D] bf16; V -> transposed [N,H,D,T] bf16.
// ---------------------------------------------------------------------------
__global__ __launch_bounds__(256) void qkv_gemm(
    const __bf16* __restrict__ xb,
    const __bf16* __restrict__ wq, const __bf16* __restrict__ wk,
    const __bf16* __restrict__ wv,
    __bf16* __restrict__ Qb, __bf16* __restrict__ Kb, __bf16* __restrict__ Vtb) {
  constexpr int LDK = 72;                        // 64 + 8 pad (keeps 16B align)
  __shared__ __bf16 Xs[2][128][LDK];             // 36 KB
  __shared__ __bf16 Ws[2][128][LDK];             // 36 KB

  const int tid  = threadIdx.x;
  const int lane = tid & 31, wave = tid >> 5;
  const int l16  = lane & 15, half = lane >> 4;
  const int wm   = wave & 3;                     // M sub-block (x32)
  const int wn   = wave >> 2;                    // N sub-block (x64)

  int bx   = blockIdx.x;                         // 3 * 64 * 6 = 1152 blocks
  int mat  = bx / (64 * 6);
  int rem  = bx % (64 * 6);
  int mblk = (rem / 6) * 128;
  int nblk = (rem % 6) * 128;
  const __bf16* W = (mat == 0) ? wq : (mat == 1) ? wk : wv;

  const int frow = tid >> 1;                     // 0..127 (fill row)
  const int fseg = (tid & 1) * 32;               // 0 / 32 (fill col base)

  auto fill = [&](int buf, int kc) {
    const __bf16* gx = xb + (size_t)(mblk + frow) * E_ + kc + fseg;
    const __bf16* gw = W  + (size_t)(nblk + frow) * E_ + kc + fseg;
    const __bf16* lx = &Xs[buf][frow][fseg];
    const __bf16* lw = &Ws[buf][frow][fseg];
#pragma unroll
    for (int i = 0; i < 4; ++i) {
      async_b128(gx + i * 8, lx + i * 8);
      async_b128(gw + i * 8, lw + i * 8);
    }
  };

  v8f acc[2][4];
#pragma unroll
  for (int s = 0; s < 2; ++s)
#pragma unroll
    for (int j = 0; j < 4; ++j) acc[s][j] = (v8f){};

  fill(0, 0);
  wait_async0();
  __syncthreads();

  for (int c = 0; c < E_ / 64; ++c) {            // 12 chunks
    int buf = c & 1;
    if (c + 1 < E_ / 64) fill(buf ^ 1, (c + 1) * 64);   // overlaps compute

#pragma unroll
    for (int ks = 0; ks < 64; ks += 32) {
      v16bf a0 = make_a_frag(&Xs[buf][wm * 32 + l16][0],      ks, half);
      v16bf a1 = make_a_frag(&Xs[buf][wm * 32 + 16 + l16][0], ks, half);
      v16bf b0 = make_b_frag(&Ws[buf][wn * 64 +  0 + l16][ks + half * 16]);
      v16bf b1 = make_b_frag(&Ws[buf][wn * 64 + 16 + l16][ks + half * 16]);
      v16bf b2 = make_b_frag(&Ws[buf][wn * 64 + 32 + l16][ks + half * 16]);
      v16bf b3 = make_b_frag(&Ws[buf][wn * 64 + 48 + l16][ks + half * 16]);
      acc[0][0] = wmma_bf16(a0, b0, acc[0][0]);
      acc[0][1] = wmma_bf16(a0, b1, acc[0][1]);
      acc[0][2] = wmma_bf16(a0, b2, acc[0][2]);
      acc[0][3] = wmma_bf16(a0, b3, acc[0][3]);
      acc[1][0] = wmma_bf16(a1, b0, acc[1][0]);
      acc[1][1] = wmma_bf16(a1, b1, acc[1][1]);
      acc[1][2] = wmma_bf16(a1, b2, acc[1][2]);
      acc[1][3] = wmma_bf16(a1, b3, acc[1][3]);
    }
    wait_async0();
    __syncthreads();
  }

  // Scatter-store (bf16). Q,K: [N,H,T,D]; V transposed: [N,H,D,T].
#pragma unroll
  for (int s = 0; s < 2; ++s) {
#pragma unroll
    for (int r = 0; r < 8; ++r) {
      int m = mblk + wm * 32 + s * 16 + r + 8 * half;
      int nb = m / T_, t = m % T_;
#pragma unroll
      for (int j = 0; j < 4; ++j) {
        int n = nblk + wn * 64 + j * 16 + l16;
        int h = n >> 6, d = n & 63;
        __bf16 bv = (__bf16)acc[s][j][r];
        if (mat == 0)      Qb [(((size_t)nb * H_ + h) * T_ + t) * D_ + d] = bv;
        else if (mat == 1) Kb [(((size_t)nb * H_ + h) * T_ + t) * D_ + d] = bv;
        else               Vtb[(((size_t)nb * H_ + h) * D_ + d) * T_ + t] = bv;
      }
    }
  }
}

// ---------------------------------------------------------------------------
// Flash attention. Block = one (n,h) x 128 query rows; wave = 16 rows.
// K (32x64) and V^T (64x32) tiles async-staged in LDS, double buffered and
// shared by all 8 waves (8x L2-traffic reduction). 4 score + 4 PV WMMAs per
// 32-key step; online softmax via 16-lane butterfly reductions.
// ---------------------------------------------------------------------------
__global__ __launch_bounds__(256) void attn(
    const __bf16* __restrict__ Qb, const __bf16* __restrict__ Kb,
    const __bf16* __restrict__ Vtb, float* __restrict__ out) {
  __shared__ __bf16 Ks[2][32][72];   // keys x d      (9 KB)
  __shared__ __bf16 Vs[2][64][40];   // d x keys      (10 KB)
  __shared__ __bf16 Ps[8][16][40];   // per-wave P    (10 KB)

  const int tid  = threadIdx.x;
  const int lane = tid & 31, wave = tid >> 5;
  const int l16  = lane & 15, half = lane >> 4;

  const int blkPerHead = T_ / 128;                  // 16
  int nh = blockIdx.x / blkPerHead;
  int tb = blockIdx.x % blkPerHead;
  int t0 = tb * 128 + wave * 16;
  int nb = nh / H_, h = nh % H_;

  const __bf16* Qh = Qb  + (size_t)nh * T_ * D_;
  const __bf16* Kh = Kb  + (size_t)nh * T_ * D_;
  const __bf16* Vh = Vtb + (size_t)nh * D_ * T_;    // [D][T]

  auto fill = [&](int buf, int s) {
    // K tile: 32 rows x 128B -> 1 b128 per thread.
    async_b128(Kh + (size_t)(s + (tid >> 3)) * D_ + (tid & 7) * 8,
               &Ks[buf][tid >> 3][(tid & 7) * 8]);
    // V tile: 64 rows x 64B -> 1 b128 per thread.
    async_b128(Vh + (size_t)(tid >> 2) * T_ + s + (tid & 3) * 8,
               &Vs[buf][tid >> 2][(tid & 3) * 8]);
  };

  // Q A-fragments kept in registers for the whole pass.
  const __bf16* qrow = Qh + (size_t)(t0 + l16) * D_;
  v16bf qa0 = make_a_frag(qrow, 0, half);
  v16bf qa1 = make_a_frag(qrow, 32, half);

  v8f o0 = {}, o1 = {}, o2 = {}, o3 = {};
  float mrow[8], lrow[8];
#pragma unroll
  for (int r = 0; r < 8; ++r) { mrow[r] = -1e30f; lrow[r] = 0.0f; }

  const float scale = 0.125f;                       // 1/sqrt(64)

  fill(0, 0);
  wait_async0();
  __syncthreads();

  for (int si = 0; si < T_ / 32; ++si) {            // 64 key tiles
    int buf = si & 1;
    if (si + 1 < T_ / 32) fill(buf ^ 1, (si + 1) * 32);   // overlaps compute

    // K^T B-fragments from LDS (16B-aligned, padded rows).
    v16bf kb00 = make_b_frag(&Ks[buf][l16][half * 16]);
    v16bf kb01 = make_b_frag(&Ks[buf][l16][32 + half * 16]);
    v16bf kb10 = make_b_frag(&Ks[buf][16 + l16][half * 16]);
    v16bf kb11 = make_b_frag(&Ks[buf][16 + l16][32 + half * 16]);

    v8f s0 = {}, s1 = {};
    s0 = wmma_bf16(qa0, kb00, s0);
    s0 = wmma_bf16(qa1, kb01, s0);
    s1 = wmma_bf16(qa0, kb10, s1);
    s1 = wmma_bf16(qa1, kb11, s1);

    // Online softmax (row = r + 8*half; 16 cols live across a 16-lane group).
#pragma unroll
    for (int r = 0; r < 8; ++r) {
      float v0 = s0[r] * scale, v1 = s1[r] * scale;
      float mx = fmaxf(v0, v1);
#pragma unroll
      for (int m = 1; m <= 8; m <<= 1) mx = fmaxf(mx, __shfl_xor(mx, m, 32));
      float mnew  = fmaxf(mrow[r], mx);
      float alpha = __expf(mrow[r] - mnew);
      float p0 = __expf(v0 - mnew);
      float p1 = __expf(v1 - mnew);
      float rs = p0 + p1;
#pragma unroll
      for (int m = 1; m <= 8; m <<= 1) rs += __shfl_xor(rs, m, 32);
      lrow[r] = lrow[r] * alpha + rs;
      mrow[r] = mnew;
      o0[r] *= alpha; o1[r] *= alpha; o2[r] *= alpha; o3[r] *= alpha;
      int prow = r + 8 * half;
      Ps[wave][prow][l16]      = (__bf16)p0;        // keys s..s+15
      Ps[wave][prow][16 + l16] = (__bf16)p1;        // keys s+16..s+31
    }

    wait_ds0();   // same-wave LDS RAW before re-reading P in A-layout

    v16bf pa = make_a_frag(&Ps[wave][l16][0], 0, half);

    v16bf vb0 = make_b_frag(&Vs[buf][ 0 + l16][half * 16]);
    v16bf vb1 = make_b_frag(&Vs[buf][16 + l16][half * 16]);
    v16bf vb2 = make_b_frag(&Vs[buf][32 + l16][half * 16]);
    v16bf vb3 = make_b_frag(&Vs[buf][48 + l16][half * 16]);
    o0 = wmma_bf16(pa, vb0, o0);
    o1 = wmma_bf16(pa, vb1, o1);
    o2 = wmma_bf16(pa, vb2, o2);
    o3 = wmma_bf16(pa, vb3, o3);

    wait_async0();          // next tile landed in LDS
    __syncthreads();        // all waves done reading this tile
  }

  // Normalize, store fp32 [N,T,E].
#pragma unroll
  for (int r = 0; r < 8; ++r) {
    float inv = 1.0f / lrow[r];
    int t = t0 + r + 8 * half;
    float* od = out + ((size_t)nb * T_ + t) * E_ + h * D_ + l16;
    od[0]  = o0[r] * inv;
    od[16] = o1[r] * inv;
    od[32] = o2[r] * inv;
    od[48] = o3[r] * inv;
  }
}

// ---------------------------------------------------------------------------
extern "C" void kernel_launch(void* const* d_in, const int* in_sizes, int n_in,
                              void* d_out, int out_size, void* d_ws, size_t ws_size,
                              hipStream_t stream) {
  (void)in_sizes; (void)n_in; (void)out_size; (void)ws_size;
  const float* x  = (const float*)d_in[0];
  const float* Wq = (const float*)d_in[1];
  const float* Wk = (const float*)d_in[2];
  const float* Wv = (const float*)d_in[3];
  float* out = (float*)d_out;

  // bf16 workspace (~54 MB)
  __bf16* ws = (__bf16*)d_ws;
  const size_t xElems = (size_t)NT * E_;
  const size_t wElems = (size_t)E_ * E_;
  const size_t qElems = (size_t)NH * T_ * D_;
  __bf16* xb  = ws;
  __bf16* wqb = xb  + xElems;
  __bf16* wkb = wqb + wElems;
  __bf16* wvb = wkb + wElems;
  __bf16* Qb  = wvb + wElems;
  __bf16* Kb  = Qb  + qElems;
  __bf16* Vtb = Kb  + qElems;

  int n4x = (int)(xElems / 4);
  cvt_f32_bf16<<<(n4x + 255) / 256, 256, 0, stream>>>(x, xb, n4x);
  int n4w = (int)(wElems / 4);
  cvt_f32_bf16<<<(n4w + 255) / 256, 256, 0, stream>>>(Wq, wqb, n4w);
  cvt_f32_bf16<<<(n4w + 255) / 256, 256, 0, stream>>>(Wk, wkb, n4w);
  cvt_f32_bf16<<<(n4w + 255) / 256, 256, 0, stream>>>(Wv, wvb, n4w);

  qkv_gemm<<<3 * (NT / 128) * (E_ / 128), 256, 0, stream>>>(xb, wqb, wkb, wvb,
                                                            Qb, Kb, Vtb);
  attn<<<NH * (T_ / 128), 256, 0, stream>>>(Qb, Kb, Vtb, out);
}
